// AttentionModel_PCA_35991825940928
// MI455X (gfx1250) — matrix operand order repaired and checked by reading, not compiled, not run on hardware
//
#include <hip/hip_runtime.h>

// MI455X / gfx1250 (wave32, WMMA). Loss kernel for the attention-PCA reference.
// Heavy stage (90 GFLOP) mapped to v_wmma_f32_16x16x32_f16; sf kept f16 in ws
// (L2-resident, 2MB); mat_ene never hits HBM (fused logsumexp epilogue).
// Gather slab double-buffered: one barrier per K-step, LDS build overlaps WMMA.

typedef __attribute__((ext_vector_type(16))) _Float16 v16h;
typedef __attribute__((ext_vector_type(8)))  _Float16 v8h;
typedef __attribute__((ext_vector_type(8)))  float    v8f;

#define H_   16
#define D_   64
#define N1_  256
#define N2_  256
#define Q1_  21
#define Q2_  21
#define M_   2048
#define LAMBD_ 0.001f

// ---------------------------------------------------------------------------
// Kernel 1: e[h,i,j] = sum_d Q[h,d,i]*K[h,d,j]; softmax over j; store sf f16.
// grid = H*16 (16 i-blocks of 16 rows), block = 256 (one thread per j).
// ---------------------------------------------------------------------------
__global__ void __launch_bounds__(256)
k_qk_softmax(const float* __restrict__ Q, const float* __restrict__ K,
             _Float16* __restrict__ sf) {
  const int h    = blockIdx.x >> 4;
  const int i0   = (blockIdx.x & 15) * 16;
  const int t    = threadIdx.x;

  __shared__ float Qs[D_][16];       // [d][ii]
  __shared__ float Es[16][N2_];      // [ii][j] logits
  __shared__ float rowmax[16], rowsum[16];

  for (int idx = t; idx < D_ * 16; idx += 256) {
    const int d = idx >> 4, ii = idx & 15;
    Qs[d][ii] = Q[(h * D_ + d) * N1_ + i0 + ii];
  }
  __syncthreads();

  // K column j = t resident in registers, reused for 16 rows.
  float kreg[D_];
  #pragma unroll
  for (int d = 0; d < D_; ++d) kreg[d] = K[(h * D_ + d) * N2_ + t];

  #pragma unroll 4
  for (int ii = 0; ii < 16; ++ii) {
    float s = 0.f;
    #pragma unroll
    for (int d = 0; d < D_; ++d) s += kreg[d] * Qs[d][ii];
    Es[ii][t] = s;
  }
  __syncthreads();

  // 8 waves; wave w reduces rows 2w, 2w+1 (max then sum of exp).
  const int wave = t >> 5, lane = t & 31;
  for (int ii = wave * 2; ii < wave * 2 + 2; ++ii) {
    float v[8], mx = -1e30f, sm = 0.f;
    #pragma unroll
    for (int c = 0; c < 8; ++c) { v[c] = Es[ii][lane + c * 32]; mx = fmaxf(mx, v[c]); }
    #pragma unroll
    for (int off = 16; off > 0; off >>= 1) mx = fmaxf(mx, __shfl_xor(mx, off, 32));
    #pragma unroll
    for (int c = 0; c < 8; ++c) sm += __expf(v[c] - mx);
    #pragma unroll
    for (int off = 16; off > 0; off >>= 1) sm += __shfl_xor(sm, off, 32);
    if (lane == 0) { rowmax[ii] = mx; rowsum[ii] = sm; }
  }
  __syncthreads();

  #pragma unroll 4
  for (int ii = 0; ii < 16; ++ii) {
    const float p = __expf(Es[ii][t] - rowmax[ii]) / rowsum[ii];
    sf[(h * N1_ + i0 + ii) * N2_ + t] = (_Float16)p;  // layout [h][i][j]
  }
}

// ---------------------------------------------------------------------------
// Kernel 2: main contraction + fused logsumexp/selection epilogue.
// grid = M/16 = 128 (m-tile), block = 512 = 16 waves (wave = 16-row i-tile).
// C_q[i,m] = sum_h sum_j sf[h][i][j] * V[h][q][Z[j,m]]   (21 f32 accumulators)
// ---------------------------------------------------------------------------
__global__ void __launch_bounds__(512)
k_main(const _Float16* __restrict__ sf, const int* __restrict__ Z,
       const float* __restrict__ V, const float* __restrict__ wts,
       float* __restrict__ pl_part) {
  const int mt   = blockIdx.x;
  const int t    = threadIdx.x;
  const int wave = t >> 5;        // i-tile index (0..15)
  const int lane = t & 31;
  const int hi   = lane >> 4;
  const int lo   = lane & 15;

  __shared__ __align__(16) _Float16 Vlds[H_ * 441 + 8];  // V as f16 [h][q][z]
  __shared__ unsigned short Zlds[N2_ * 16];              // Z tile [j][m]
  __shared__ __align__(16) _Float16 Gs[2][Q1_][16][32];  // double-buffered B slab
  __shared__ float wgred[16];

  for (int idx = t; idx < H_ * 441; idx += 512) Vlds[idx] = (_Float16)V[idx];
  for (int idx = t; idx < N2_ * 16; idx += 512) {
    const int j = idx >> 4, m = idx & 15;
    Zlds[idx] = (unsigned short)Z[j * M_ + mt * 16 + m];
  }
  __syncthreads();

  // Slab builder: thread t owns one (j,m) pair (32*16 = 512 pairs) per step.
  const int bj = t >> 4, bm = t & 15;

  // Prologue: build slab for step 0 (h=0, jc=0) into buffer 0.
  {
    const int s = Zlds[bj * 16 + bm];
    const _Float16* vcol = &Vlds[0 * 441 + s];
    #pragma unroll
    for (int q = 0; q < Q1_; ++q) Gs[0][q][bm][bj] = vcol[q * Q2_];
  }

  v8f acc[Q1_];
  const v8f vzero = {0.f, 0.f, 0.f, 0.f, 0.f, 0.f, 0.f, 0.f};
  #pragma unroll
  for (int q = 0; q < Q1_; ++q) acc[q] = vzero;

  // 128 K-steps: step = h*8 + jc, each contracts 32 j's of one head.
  for (int it = 0; it < H_ * (N2_ / 32); ++it) {
    const int h  = it >> 3;
    const int jc = it & 7;
    const int p  = it & 1;

    __syncthreads();  // buf p fully built; buf p^1 no longer read

    // A tile (16x32 f16), ISA layout: lane row = lo; halves 0-7 -> K=hi*8+0..7,
    // halves 8-15 -> K=16+hi*8+0..7.
    const _Float16* arow = sf + ((h * N1_ + wave * 16 + lo) * N2_) + jc * 32 + hi * 8;
    union { v16h v; v8h pr[2]; } A;
    A.pr[0] = *(const v8h*)(arow);
    A.pr[1] = *(const v8h*)(arow + 16);
    if (it + 1 < H_ * (N2_ / 32)) {
      const int hn  = (it + 1) >> 3;
      const int jcn = (it + 1) & 7;
      __builtin_prefetch(sf + ((hn * N1_ + wave * 16 + lo) * N2_) + jcn * 32, 0, 0);
    }

    #pragma unroll
    for (int q = 0; q < Q1_; ++q) {
      // B tile (32x16 f16): lane col = lo, K = hi*8 + 0..7, 16 + hi*8 + 0..7.
      union { v16h v; v8h pr[2]; } bu;
      bu.pr[0] = *(const v8h*)(&Gs[p][q][lo][hi * 8]);
      bu.pr[1] = *(const v8h*)(&Gs[p][q][lo][16 + hi * 8]);
      const v16h b = bu.v;
      acc[q] = __builtin_amdgcn_wmma_f32_16x16x32_f16(
          false, A.v, false, b, (short)0, acc[q], false, false);
    }

    // Build next step's slab into the other buffer; these ds_stores overlap
    // with the WMMAs above (issued after the B ds_loads, same-wave in-order).
    if (it + 1 < H_ * (N2_ / 32)) {
      const int hn  = (it + 1) >> 3;
      const int jcn = (it + 1) & 7;
      const int s   = Zlds[(jcn * 32 + bj) * 16 + bm];
      const _Float16* vcol = &Vlds[hn * 441 + s];
      #pragma unroll
      for (int q = 0; q < Q1_; ++q) Gs[p ^ 1][q][bm][bj] = vcol[q * Q2_];
    }
  }
  __syncthreads();

  // Fused epilogue: lge over q, select q = Z[i,m], weighted partial sum.
  const float wm = wts[mt * 16 + lo];
  float part = 0.f;
  #pragma unroll
  for (int r = 0; r < 8; ++r) {
    const int irow = wave * 16 + r + 8 * hi;  // C/D layout: M = r + 8*hi, N = lo
    float cmax = -1e30f;
    #pragma unroll
    for (int q = 0; q < Q1_; ++q) cmax = fmaxf(cmax, acc[q][r]);
    float s = 0.f;
    #pragma unroll
    for (int q = 0; q < Q1_; ++q) s += __expf(acc[q][r] - cmax);
    const float lge = __logf(s) + cmax;
    const int z = (int)Zlds[irow * 16 + lo];
    float sel = 0.f;
    #pragma unroll
    for (int q = 0; q < Q1_; ++q) sel = (q == z) ? acc[q][r] : sel;
    part += wm * (sel - lge);
  }
  #pragma unroll
  for (int off = 16; off > 0; off >>= 1) part += __shfl_xor(part, off, 32);
  if (lane == 0) wgred[wave] = part;
  __syncthreads();
  if (t == 0) {
    float s = 0.f;
    for (int i = 0; i < 16; ++i) s += wgred[i];
    pl_part[mt] = s;
  }
}

// ---------------------------------------------------------------------------
// Kernel 3: Mm partials = sf^T sf (16x16 over ij=65536) via WMMA.
// grid = 64 single-wave WGs; each covers 1024 of the ij range.
// ---------------------------------------------------------------------------
__global__ void __launch_bounds__(32)
k_gram(const _Float16* __restrict__ sf, float* __restrict__ mm_part) {
  const int wg   = blockIdx.x;
  const int lane = threadIdx.x;
  const int hi   = lane >> 4, lo = lane & 15;

  v8f acc = {0.f, 0.f, 0.f, 0.f, 0.f, 0.f, 0.f, 0.f};
  for (int c = 0; c < 32; ++c) {
    const int base = wg * 1024 + c * 32;
    // A[m=h][k] = sf[h][base+k] (interleaved-K A layout)
    const _Float16* arow = sf + lo * (N1_ * N2_) + base + hi * 8;
    union { v16h v; v8h p[2]; } A;
    A.p[0] = *(const v8h*)(arow);
    A.p[1] = *(const v8h*)(arow + 16);
    // B[k][n=h'] = sf[h'][base+k] (interleaved-K B layout)
    union { v16h v; v8h p[2]; } Bu;
    Bu.p[0] = *(const v8h*)(sf + lo * (N1_ * N2_) + base + hi * 8);
    Bu.p[1] = *(const v8h*)(sf + lo * (N1_ * N2_) + base + hi * 8 + 16);
    const v16h B = Bu.v;
    acc = __builtin_amdgcn_wmma_f32_16x16x32_f16(
        false, A.v, false, B, (short)0, acc, false, false);
  }
  #pragma unroll
  for (int r = 0; r < 8; ++r) {
    const int m = r + 8 * hi;
    mm_part[wg * 256 + m * 16 + lo] = acc[r];
  }
}

// ---------------------------------------------------------------------------
// Kernel 4: deterministic final reduce: -sum(pl) + LAMBD*sum(Mm .* (VV VV^T)).
// ---------------------------------------------------------------------------
__global__ void __launch_bounds__(256)
k_final(const float* __restrict__ pl_part, const float* __restrict__ mm_part,
        const float* __restrict__ V, float* __restrict__ out) {
  __shared__ float red[256];
  const int t = threadIdx.x;
  const int h = t >> 4, k = t & 15;

  float mm = 0.f;
  for (int w = 0; w < 64; ++w) mm += mm_part[w * 256 + t];
  float vv = 0.f;
  for (int n = 0; n < Q1_ * Q2_; ++n) vv += V[h * 441 + n] * V[k * 441 + n];
  red[t] = mm * vv;
  __syncthreads();
  for (int off = 128; off > 0; off >>= 1) {
    if (t < off) red[t] += red[t + off];
    __syncthreads();
  }
  if (t == 0) {
    float pl = 0.f;
    for (int i = 0; i < M_ / 16; ++i) pl += pl_part[i];
    out[0] = -pl + LAMBD_ * red[0];
  }
}

// ---------------------------------------------------------------------------
extern "C" void kernel_launch(void* const* d_in, const int* in_sizes, int n_in,
                              void* d_out, int out_size, void* d_ws, size_t ws_size,
                              hipStream_t stream) {
  (void)in_sizes; (void)n_in; (void)out_size; (void)ws_size;
  const int*   Z   = (const int*)  d_in[0];   // (N2, M) int32
  const float* wts = (const float*)d_in[1];   // (M,)
  const float* Q   = (const float*)d_in[2];   // (H, D, N1)
  const float* K   = (const float*)d_in[3];   // (H, D, N2)
  const float* V   = (const float*)d_in[4];   // (H, Q1, Q2)
  float* out = (float*)d_out;

  char* ws = (char*)d_ws;
  _Float16* sf      = (_Float16*)(ws);                               // 2 MB
  float*    pl_part = (float*)(ws + (size_t)2 * 1024 * 1024);        // 128 f32
  float*    mm_part = (float*)(ws + (size_t)2 * 1024 * 1024 + 1024); // 64*256 f32

  k_qk_softmax<<<dim3(H_ * (N1_ / 16)), dim3(256), 0, stream>>>(Q, K, sf);
  k_main      <<<dim3(M_ / 16),         dim3(512), 0, stream>>>(sf, Z, V, wts, pl_part);
  k_gram      <<<dim3(64),              dim3(32),  0, stream>>>(sf, mm_part);
  k_final     <<<dim3(1),               dim3(256), 0, stream>>>(pl_part, mm_part, V, out);
}
